// GraphAttention_59803124629978
// MI455X (gfx1250) — matrix operand order repaired and checked by reading, not compiled, not run on hardware
//
#include <hip/hip_runtime.h>
#include <hip/hip_bf16.h>

typedef __attribute__((ext_vector_type(16))) __bf16 v16bf;
typedef __attribute__((ext_vector_type(8)))  float  v8f;

#define Vt   10000
#define Dm   300
#define Bt   64
#define Lt   128
#define CF   5.0f
#define EPSm 1e-8f
#define NT   20      // 320 padded cols / 16
#define NW   5       // waves per block in main kernel
#define TPW  4       // n-tiles per wave (5*4 = 20)
#define NCH  313     // ceil(10000/32) k-chunks
#define VG   8       // v-dimension block split
#define CPG  40      // chunks per group (ceil(313/8))

// ---------------- zero output ----------------
__global__ void k_zero(float* p, int n) {
    int i = blockIdx.x * blockDim.x + threadIdx.x;
    if (i < n) p[i] = 0.0f;
}

// ---------------- per-sample masked mean [B,300] ----------------
__global__ __launch_bounds__(320) void k_persample(const int* cl, const int* clen,
                                                   const float* embed, float* ps) {
    int b = blockIdx.x, d = threadIdx.x;
    if (d >= Dm) return;
    int len = clen[b];
    float denom = (float)(len > 0 ? len : 1);
    float s = 0.0f;
    for (int l = 0; l < len; ++l) {
        int idx = cl[b * Lt + l];
        s += embed[(size_t)idx * Dm + d];
    }
    ps[b * Dm + d] = s / denom;
}

// ---------------- context + ||context|| ----------------
__global__ __launch_bounds__(320) void k_context(const float* ps, float* ctx, float* cn) {
    int d = threadIdx.x;
    float c = 0.0f;
    if (d < Dm) {
        for (int b = 0; b < Bt; ++b) c += ps[b * Dm + d];
        c *= (1.0f / Bt);
        ctx[d] = c;
    }
    __shared__ float red[320];
    red[d] = (d < Dm) ? c * c : 0.0f;
    __syncthreads();
    for (int st = 256; st > 0; st >>= 1) {
        if (d < st && d + st < 320) red[d] += red[d + st];
        __syncthreads();
    }
    if (d == 0) cn[0] = sqrtf(red[0]);
}

// ---------------- |cosine| per concept row (one wave32 per row) ----------------
__global__ __launch_bounds__(256) void k_cos(const float* cw, const float* ctx,
                                             const float* cn, float* cosv) {
    int wave = threadIdx.x >> 5, lane = threadIdx.x & 31;
    int v = blockIdx.x * 8 + wave;
    if (v >= Vt) return;
    float dot = 0.0f, nr = 0.0f;
    for (int k = lane; k < Dm; k += 32) {
        float x = cw[(size_t)v * Dm + k];
        dot += x * ctx[k];
        nr  += x * x;
    }
    for (int o = 16; o > 0; o >>= 1) {
        dot += __shfl_xor(dot, o, 32);
        nr  += __shfl_xor(nr,  o, 32);
    }
    if (lane == 0)
        cosv[v] = fabsf(dot) / fmaxf(cn[0] * sqrtf(nr), EPSm);
}

// ------- pack concept_w into bf16 WMMA B-fragments: [chunk][tile][lane][elem] -------
// B (K=32 x N=16) layout: lane = n + 16*((K>>3)&1), element e = (K&7) + 8*(K>>4)
__global__ void k_packB(const float* cw, __bf16* cwB, int total) {
    int x = blockIdx.x * blockDim.x + threadIdx.x;
    if (x >= total) return;
    int e = x & 15;
    int lane = (x >> 4) & 31;
    int rest = x >> 9;
    int t = rest % NT, ch = rest / NT;
    int k = ch * 32 + (e & 7) + ((lane >= 16) ? 8 : 0) + ((e >> 3) << 4);
    int col = t * 16 + (lane & 15);
    float val = (k < Vt && col < Dm) ? cw[(size_t)k * Dm + col] : 0.0f;
    cwB[x] = (__bf16)val;
}

__device__ __forceinline__ float edge_weight(float em, float la, float cs, float af) {
    return CF * (la * em * cs + (1.0f - la) * ((em > 0.0f) ? af : 0.0f));
}

// ---------------- pass 1: per-row softmax max & sum(exp) ----------------
__global__ __launch_bounds__(256) void k_rowstats(const int* cl, const int* clen,
                                                  const float* edge, const float* aff,
                                                  const float* lam, const float* cosv,
                                                  float* rowmax, float* rowsum) {
    int r = blockIdx.x;
    int b = r >> 7, l = r & (Lt - 1);
    int tid = threadIdx.x;
    if (l >= clen[b]) {
        if (tid == 0) { rowmax[r] = 0.0f; rowsum[r] = 1.0f; }
        return;
    }
    size_t base = (size_t)cl[r] * Vt;
    float m = -1e30f, s = 0.0f;
    for (int v = tid; v < Vt; v += 256) {
        float w = edge_weight(edge[base + v], lam[v], cosv[v], aff[v]);
        if (w > m) { s *= __expf(m - w); m = w; }
        s += __expf(w - m);
    }
    __shared__ float ms[256], ss[256];
    ms[tid] = m; ss[tid] = s;
    __syncthreads();
    for (int st = 128; st > 0; st >>= 1) {
        if (tid < st) {
            float m1 = ms[tid], m2 = ms[tid + st];
            float mn = fmaxf(m1, m2);
            ss[tid] = ss[tid] * __expf(m1 - mn) + ss[tid + st] * __expf(m2 - mn);
            ms[tid] = mn;
        }
        __syncthreads();
    }
    if (tid == 0) { rowmax[r] = ms[0]; rowsum[r] = ss[0]; }
}

// ---------------- pass 2: fused probs + WMMA matmul + masked-mean reduce ----------------
__global__ __launch_bounds__(160) void k_main(const int* cl, const int* clen,
                                              const float* edge, const float* aff,
                                              const float* lam, const float* cosv,
                                              const float* rowmax, const float* rowsum,
                                              const v16bf* cwB, float* out) {
    int gid = blockIdx.x;
    int vg = gid % VG;
    int lt = (gid / VG) & 7;
    int b  = gid / (VG * 8);
    int len = clen[b];
    if (lt * 16 >= len) return;   // fully-masked tile: contributes zero

    __shared__ __bf16 smA[32 * 16];
    __shared__ int    sidx[16];
    __shared__ float  scoef[16];
    __shared__ float  smax[16];

    int tid = threadIdx.x;
    if (tid < 16) {
        int l = lt * 16 + tid;
        int r = b * Lt + l;
        sidx[tid] = cl[r];
        smax[tid] = rowmax[r];
        float dn = (float)(len > 0 ? len : 1);
        scoef[tid] = (l < len) ? 1.0f / (dn * rowsum[r]) : 0.0f;
    }
    __syncthreads();

    int lane = tid & 31;
    int wv   = tid >> 5;       // 0..4
    v8f acc[TPW];
#pragma unroll
    for (int t = 0; t < TPW; ++t)
#pragma unroll
        for (int i = 0; i < 8; ++i) acc[t][i] = 0.0f;

    int c0 = vg * CPG;
    int c1 = (c0 + CPG < NCH) ? c0 + CPG : NCH;
    for (int c = c0; c < c1; ++c) {
        // build P chunk (16 rows x 32 k) directly into A-fragment LDS layout
        for (int i = tid; i < 512; i += 160) {
            int rr = i >> 5, j = i & 31;
            int v = c * 32 + j;
            float p = 0.0f;
            if (v < Vt) {
                float em = edge[(size_t)sidx[rr] * Vt + v];
                float w = edge_weight(em, lam[v], cosv[v], aff[v]);
                p = __expf(w - smax[rr]);
            }
            int ln = rr + (((j >> 3) & 1) << 4);       // A layout: lane half from K bit3
            int e  = (j & 7) + ((j >> 4) << 3);        // element from K bits {2:0,4}
            smA[ln * 16 + e] = (__bf16)p;
        }
        __syncthreads();

        v16bf a = *reinterpret_cast<const v16bf*>(&smA[lane * 16]);
#pragma unroll
        for (int t = 0; t < TPW; ++t) {
            int tg = wv * TPW + t;
            v16bf bm = cwB[(size_t)(c * NT + tg) * 32 + lane];
            acc[t] = __builtin_amdgcn_wmma_f32_16x16x32_bf16(
                false, a, false, bm, (short)0, acc[t], false, false);
        }
        __syncthreads();
    }

    // epilogue: C layout — VGPR i, lanes 0-15 -> row i, lanes 16-31 -> row i+8; col = lane&15
    int hi = (lane >> 4) << 3;
#pragma unroll
    for (int t = 0; t < TPW; ++t) {
        float part = 0.0f;
#pragma unroll
        for (int i = 0; i < 8; ++i) part += acc[t][i] * scoef[i + hi];
        part += __shfl_down(part, 16, 32);             // fold upper lane-half rows
        int col = (wv * TPW + t) * 16 + (lane & 15);
        if (lane < 16 && col < Dm)
            atomicAdd(&out[b * Dm + col], part);
    }
}

extern "C" void kernel_launch(void* const* d_in, const int* in_sizes, int n_in,
                              void* d_out, int out_size, void* d_ws, size_t ws_size,
                              hipStream_t stream) {
    (void)in_sizes; (void)n_in; (void)out_size; (void)ws_size;
    const int*   cl    = (const int*)d_in[0];
    const int*   clen  = (const int*)d_in[1];
    const float* embed = (const float*)d_in[2];
    const float* cw    = (const float*)d_in[3];
    const float* edge  = (const float*)d_in[4];
    const float* aff   = (const float*)d_in[5];
    const float* lam   = (const float*)d_in[6];
    float* out = (float*)d_out;

    char* ws = (char*)d_ws;
    float* ps   = (float*)(ws);              // 64*300 f32
    float* ctx  = (float*)(ws + 76800);      // 300 f32
    float* cn   = (float*)(ws + 78000);      // 1 f32 (padded)
    float* cosv = (float*)(ws + 78016);      // 10000 f32
    float* rmax = (float*)(ws + 118016);     // 8192 f32
    float* rsum = (float*)(ws + 150784);     // 8192 f32
    v16bf* cwB  = (v16bf*)(ws + 183616);     // 313*20*32 fragments (6.4 MB), 32B aligned

    const int npack = NCH * NT * 32 * 16;    // 3,205,120 bf16 elements

    k_zero     <<<(Bt * Dm + 255) / 256, 256, 0, stream>>>(out, Bt * Dm);
    k_persample<<<Bt, 320, 0, stream>>>(cl, clen, embed, ps);
    k_context  <<<1, 320, 0, stream>>>(ps, ctx, cn);
    k_cos      <<<(Vt + 7) / 8, 256, 0, stream>>>(cw, ctx, cn, cosv);
    k_packB    <<<(npack + 255) / 256, 256, 0, stream>>>(cw, (__bf16*)cwB, npack);
    k_rowstats <<<Bt * Lt, 256, 0, stream>>>(cl, clen, edge, aff, lam, cosv, rmax, rsum);
    k_main     <<<Bt * 8 * VG, NW * 32, 0, stream>>>(cl, clen, edge, aff, lam, cosv,
                                                     rmax, rsum, cwB, out);
}
